// LePEAttention_13391708029361
// MI455X (gfx1250) — matrix operand order repaired and checked by reading, not compiled, not run on hardware
//
#include <hip/hip_runtime.h>

typedef __attribute__((ext_vector_type(16))) _Float16 v16h;
typedef __attribute__((ext_vector_type(8)))  _Float16 v8h;
typedef __attribute__((ext_vector_type(2)))  _Float16 h2;
typedef __attribute__((ext_vector_type(8)))  float    v8f;

#define DIM     512
#define HEADS   8
#define HD      64
#define NB      16
#define WIMG    56
#define HIMG    56
#define LTOK    (HIMG*WIMG)      /* 3136 */
#define WSP     7
#define HSP     56
#define SSEQ    (HSP*WSP)        /* 392 */
#define NWIN_W  (WIMG/WSP)       /* 8   */
#define SPAD    416              /* 13*32, pad of 392 */
#define KSTEP   32
#define NKIT    (SPAD/KSTEP)     /* 13  */
#define NWAVE   25               /* 25*16 = 400 >= 392 query rows */
#define BLOCK_T (NWAVE*32)       /* 800 threads */

union V16U { v16h v; v8h h[2]; _Float16 e[16]; };

__device__ __forceinline__ v8f wmma_f16(v16h a, v16h b, v8f c) {
  // 8 args: (neg_a, A, neg_b, B, c_mod, C, reuse_a, reuse_b)
  return __builtin_amdgcn_wmma_f32_16x16x32_f16(false, a, false, b, (short)0, c,
                                                false, false);
}

// compare-select max: avoids v_max_num canonicalization (inputs are never NaN)
__device__ __forceinline__ float tmax(float a, float b) { return a > b ? a : b; }

__device__ __forceinline__ h2 shfl_xor_h2(h2 x, int off) {
  int xi = __builtin_bit_cast(int, x);
  xi = __shfl_xor(xi, off, 32);
  return __builtin_bit_cast(h2, xi);
}

// packed f16 max via VOP3P V_PK_MAX_NUM_F16 (one instruction per butterfly level)
__device__ __forceinline__ h2 pkmax(h2 a, h2 b) {
  h2 d;
  asm("v_pk_max_num_f16 %0, %1, %2" : "=v"(d) : "v"(a), "v"(b));
  return d;
}

// One 32-key flash-attention step. MASK=true only for the final (padded) step.
template <bool MASK>
__device__ __forceinline__ void attn_step(
    int t, int half, int lcol,
    const _Float16* Klds, const _Float16* VTlds, _Float16* ps,
    v16h qa0, v16h qa1, v16h onesB,
    v8f (&acc)[4], float (&m_i)[8], float (&l_i)[8]) {
  const v8f zf = {0.f, 0.f, 0.f, 0.f, 0.f, 0.f, 0.f, 0.f};

  // ---- S = Q K^T for 32 keys (two 16-col tiles) ----
  const _Float16* kr0 = &Klds[(t + lcol) * HD];
  const _Float16* kr1 = &Klds[(t + 16 + lcol) * HD];
  v16h kb0lo = *(const v16h*)(kr0 + 16 * half);
  v16h kb0hi = *(const v16h*)(kr0 + 32 + 16 * half);
  v16h kb1lo = *(const v16h*)(kr1 + 16 * half);
  v16h kb1hi = *(const v16h*)(kr1 + 32 + 16 * half);

  v8f s0 = zf, s1 = zf;
  s0 = wmma_f16(qa0, kb0lo, s0);
  s0 = wmma_f16(qa1, kb0hi, s0);
  s1 = wmma_f16(qa0, kb1lo, s1);
  s1 = wmma_f16(qa1, kb1hi, s1);

  const bool ok0 = !MASK || ((t + lcol) < SSEQ);
  const bool ok1 = !MASK || ((t + 16 + lcol) < SSEQ);

  // ---- packed-f16 row-max butterfly over the 16 lanes of each half ----
  float mnew[8];
#pragma unroll
  for (int rp = 0; rp < 4; ++rp) {
    float c0, c1;
    if (MASK) {
      float a0 = ok0 ? s0[2 * rp] : -3.0e38f;
      float b0 = ok1 ? s1[2 * rp] : -3.0e38f;
      float a1 = ok0 ? s0[2 * rp + 1] : -3.0e38f;
      float b1 = ok1 ? s1[2 * rp + 1] : -3.0e38f;
      c0 = tmax(a0, b0);
      c1 = tmax(a1, b1);
    } else {
      c0 = tmax(s0[2 * rp], s1[2 * rp]);
      c1 = tmax(s0[2 * rp + 1], s1[2 * rp + 1]);
    }
    h2 mp = {(_Float16)c0, (_Float16)c1};
#pragma unroll
    for (int off = 1; off < 16; off <<= 1)
      mp = pkmax(mp, shfl_xor_h2(mp, off));
    mnew[2 * rp]     = tmax(m_i[2 * rp], (float)mp[0]);
    mnew[2 * rp + 1] = tmax(m_i[2 * rp + 1], (float)mp[1]);
  }

  // ---- exp2-domain probabilities + accumulator rescale ----
  float alpha[8];
#pragma unroll
  for (int r = 0; r < 8; ++r) {
    alpha[r] = __builtin_amdgcn_exp2f(m_i[r] - mnew[r]);
    m_i[r] = mnew[r];
    float p0 = __builtin_amdgcn_exp2f(s0[r] - mnew[r]);
    float p1 = __builtin_amdgcn_exp2f(s1[r] - mnew[r]);
    if (MASK) { p0 = ok0 ? p0 : 0.f; p1 = ok1 ? p1 : 0.f; }
    s0[r] = p0; s1[r] = p1;
#pragma unroll
    for (int c = 0; c < 4; ++c) acc[c][r] *= alpha[r];
  }

  // ---- P (C layout) -> per-wave LDS scratch as row-major 16x32 f16 ----
#pragma unroll
  for (int r = 0; r < 8; ++r) {
    int prow = r + 8 * half;
    ps[prow * 32 + lcol]      = (_Float16)s0[r];
    ps[prow * 32 + 16 + lcol] = (_Float16)s1[r];
  }
  // reload P in A-operand layout: row lcol, chunks at key 16*half and 16*half+8
  v16h pa;
  {
    const _Float16* pr = ps + lcol * 32;
    V16U u;
    u.h[0] = *(const v8h*)(pr + 16 * half);
    u.h[1] = *(const v8h*)(pr + 16 * half + 8);
    pa = u.v;
  }

  // ---- row sums via WMMA against an all-ones 32x16 B matrix ----
  v8f rs = wmma_f16(pa, onesB, zf);
#pragma unroll
  for (int r = 0; r < 8; ++r) l_i[r] = l_i[r] * alpha[r] + rs[r];

  // ---- O += P @ V ----
#pragma unroll
  for (int c = 0; c < 4; ++c) {
    const _Float16* vtr = &VTlds[(16 * c + lcol) * SPAD + t + 16 * half];
    v16h vb = *(const v16h*)vtr;
    acc[c] = wmma_f16(pa, vb, acc[c]);
  }
}

__global__ __launch_bounds__(BLOCK_T)
void lepe_attn_kernel(const float* __restrict__ q, const float* __restrict__ k,
                      const float* __restrict__ v, float* __restrict__ out) {
  __shared__ __attribute__((aligned(32))) _Float16 Klds[SPAD * HD];   // [key][d]
  __shared__ __attribute__((aligned(32))) _Float16 VTlds[HD * SPAD];  // [d][key]
  __shared__ __attribute__((aligned(32))) _Float16 Pscr[NWAVE * 16 * 32];

  const int wh   = blockIdx.x;            // 0..1023 = (window, head)
  const int head = wh % HEADS;
  const int win  = wh / HEADS;
  const int bb   = win / NWIN_W;
  const int ww   = win % NWIN_W;
  const int ch0  = head * HD;
  const int tid  = threadIdx.x;

  // ---- stage K (row-major) and V^T into LDS, f32 -> f16, zero-pad keys >= 392
  for (int i = tid; i < SPAD * HD; i += BLOCK_T) {
    int t = i >> 6, d = i & 63;
    _Float16 kv = (_Float16)0.f, vv = (_Float16)0.f;
    if (t < SSEQ) {
      int y = t / WSP, x = t - y * WSP;
      size_t g = ((size_t)(bb * LTOK + y * WIMG + ww * WSP + x)) * DIM + ch0 + d;
      kv = (_Float16)k[g];
      vv = (_Float16)v[g];
    }
    Klds[t * HD + d]    = kv;
    VTlds[d * SPAD + t] = vv;
  }
  __syncthreads();

  const int lane  = tid & 31;
  const int half  = lane >> 4;   // k-half selector for WMMA operand layouts
  const int lcol  = lane & 15;   // row (A) / column (B,C,D) index
  const int wv    = tid >> 5;    // wave id == query tile id (0..24)
  const int qbase = wv * 16;
  // fold log2(e) into scale: softmax computed in exp2 domain
  const float scale = 0.125f * 1.44269504088896340736f;

  // ---- load Q tile directly from global into two A operands (16x32 f16 each)
  int srow = qbase + lcol; if (srow >= SSEQ) srow = SSEQ - 1;
  int qy = srow / WSP, qx = srow - qy * WSP;
  const float* qrow = q + ((size_t)(bb * LTOK + qy * WIMG + ww * WSP + qx)) * DIM + ch0;
  v16h qa0, qa1;
  {
    V16U u0, u1;
    const int kb = 16 * half;
#pragma unroll
    for (int j = 0; j < 8; ++j) {
      u0.e[j]     = (_Float16)(scale * qrow[kb + j]);
      u0.e[8 + j] = (_Float16)(scale * qrow[8 + kb + j]);
      u1.e[j]     = (_Float16)(scale * qrow[32 + kb + j]);
      u1.e[8 + j] = (_Float16)(scale * qrow[40 + kb + j]);
    }
    qa0 = u0.v; qa1 = u1.v;
  }

  v16h onesB;
  {
    V16U u;
#pragma unroll
    for (int j = 0; j < 16; ++j) u.e[j] = (_Float16)1.f;
    onesB = u.v;
  }

  const v8f zf = {0.f, 0.f, 0.f, 0.f, 0.f, 0.f, 0.f, 0.f};
  v8f acc[4];
#pragma unroll
  for (int c = 0; c < 4; ++c) acc[c] = zf;
  float m_i[8], l_i[8];
#pragma unroll
  for (int r = 0; r < 8; ++r) { m_i[r] = -1.0e30f; l_i[r] = 0.f; }

  _Float16* ps = &Pscr[wv * 16 * 32];

  // ---- main loop: steps 0..11 are mask-free; step 12 handles the padded keys
  for (int it = 0; it < NKIT - 1; ++it)
    attn_step<false>(it * KSTEP, half, lcol, Klds, VTlds, ps,
                     qa0, qa1, onesB, acc, m_i, l_i);
  attn_step<true>((NKIT - 1) * KSTEP, half, lcol, Klds, VTlds, ps,
                  qa0, qa1, onesB, acc, m_i, l_i);

  // ---- epilogue: normalize by row sum, store (C layout: row=r+8*half, col=lcol)
#pragma unroll
  for (int r = 0; r < 8; ++r) {
    int sr = qbase + r + 8 * half;
    if (sr < SSEQ) {
      float inv = 1.0f / l_i[r];
      int y = sr / WSP, x = sr - y * WSP;
      float* orow = out + ((size_t)(bb * LTOK + y * WIMG + ww * WSP + x)) * DIM + ch0;
#pragma unroll
      for (int c = 0; c < 4; ++c) orow[16 * c + lcol] = acc[c][r] * inv;
    }
  }
}

// LePE: out += depthwise 3x3 cross-correlation of v over each (56 x 7) window image
__global__ __launch_bounds__(256)
void lepe_conv_kernel(const float* __restrict__ v, const float* __restrict__ w,
                      const float* __restrict__ bias, float* __restrict__ out) {
  size_t idx = (size_t)blockIdx.x * 256 + threadIdx.x;
  if (idx >= (size_t)NB * LTOK * DIM) return;
  int c = (int)(idx % DIM);
  size_t bl = idx / DIM;
  int l = (int)(bl % LTOK);
  int b = (int)(bl / LTOK);
  int yi = l / WIMG, xi = l - yi * WIMG;
  int wwin = xi / WSP, x = xi - wwin * WSP;
  float acc = bias[c];
#pragma unroll
  for (int dy = -1; dy <= 1; ++dy) {
    int y2 = yi + dy;
    if ((unsigned)y2 >= (unsigned)HSP) continue;
#pragma unroll
    for (int dx = -1; dx <= 1; ++dx) {
      int x2 = x + dx;
      if ((unsigned)x2 >= (unsigned)WSP) continue;   // window-local zero padding
      size_t g = ((size_t)(b * LTOK + y2 * WIMG + wwin * WSP + x2)) * DIM + c;
      acc = fmaf(w[c * 9 + (dy + 1) * 3 + (dx + 1)], v[g], acc);
    }
  }
  out[idx] += acc;
}

extern "C" void kernel_launch(void* const* d_in, const int* in_sizes, int n_in,
                              void* d_out, int out_size, void* d_ws, size_t ws_size,
                              hipStream_t stream) {
  (void)in_sizes; (void)n_in; (void)d_ws; (void)ws_size; (void)out_size;
  const float* qkv   = (const float*)d_in[0];
  const float* convw = (const float*)d_in[1];
  const float* convb = (const float*)d_in[2];
  const size_t blc = (size_t)NB * LTOK * DIM;
  const float* q = qkv;
  const float* k = qkv + blc;
  const float* v = qkv + 2 * blc;
  float* out = (float*)d_out;

  lepe_attn_kernel<<<NB * NWIN_W * HEADS, BLOCK_T, 0, stream>>>(q, k, v, out);

  int blocks = (int)((blc + 255) / 256);
  lepe_conv_kernel<<<blocks, 256, 0, stream>>>(v, convw, convb, out);
}
